// GraphGNN_87574383165968
// MI455X (gfx1250) — matrix operand order repaired and checked by reading, not compiled, not run on hardware
//
#include <hip/hip_runtime.h>
#include <hip/hip_bf16.h>
#include <math.h>

typedef __attribute__((ext_vector_type(16))) _Float16 v16h;
typedef __attribute__((ext_vector_type(8)))  _Float16 v8h;
typedef __attribute__((ext_vector_type(8)))  float    v8f;

#define NBATCH 8
#define NNODES 10000
#define NEDGES 320000
#define INDIM  16
#define EH     32      // hidden
#define EOUT   22      // edge MLP out
#define NOUT   16      // node MLP out

#define FS 72          // feat row stride (halves), 64 used, padded for banks
#define HS 40          // h row stride (halves), 32 used
#define NTILES 8       // 16-edge tiles per block (amortize weight staging)

// fast sigmoid: v_exp + v_rcp, no IEEE division sequence
__device__ __forceinline__ float sigmoidf_fast(float x) {
    return __builtin_amdgcn_rcpf(1.0f + __expf(-x));
}

// ---------------------------------------------------------------------------
// Kernel 0: zero agg workspace, seed min/max (positive floats -> uint order)
// ---------------------------------------------------------------------------
__global__ void k_init(unsigned* __restrict__ mm, float* __restrict__ agg, int aggN) {
    int i = blockIdx.x * blockDim.x + threadIdx.x;
    if (i == 0) {
        mm[0] = 0x7f800000u;  // min dist  (+inf)
        mm[1] = 0x7f800000u;  // min direc (+inf)
        mm[2] = 0u;           // max dist  (0.0, all data >= 1)
        mm[3] = 0u;           // max direc (0.0, all data >= 0)
    }
    for (int j = i; j < aggN; j += gridDim.x * blockDim.x) agg[j] = 0.0f;
}

// ---------------------------------------------------------------------------
// Kernel 1: column-wise min/max of edge_attr (E,2)
// ---------------------------------------------------------------------------
__global__ void k_minmax(const float* __restrict__ ea, unsigned* __restrict__ mm, int E) {
    __shared__ float s0[256], s1[256], s2[256], s3[256];
    float mn0 = INFINITY, mn1 = INFINITY, mx0 = 0.0f, mx1 = 0.0f;
    for (int e = blockIdx.x * blockDim.x + threadIdx.x; e < E;
         e += gridDim.x * blockDim.x) {
        float a = ea[2 * e + 0];
        float b = ea[2 * e + 1];
        mn0 = fminf(mn0, a); mx0 = fmaxf(mx0, a);
        mn1 = fminf(mn1, b); mx1 = fmaxf(mx1, b);
    }
    int t = threadIdx.x;
    s0[t] = mn0; s1[t] = mn1; s2[t] = mx0; s3[t] = mx1;
    __syncthreads();
    for (int stride = 128; stride > 0; stride >>= 1) {
        if (t < stride) {
            s0[t] = fminf(s0[t], s0[t + stride]);
            s1[t] = fminf(s1[t], s1[t + stride]);
            s2[t] = fmaxf(s2[t], s2[t + stride]);
            s3[t] = fmaxf(s3[t], s3[t + stride]);
        }
        __syncthreads();
    }
    if (t == 0) {
        atomicMin(&mm[0], __float_as_uint(s0[0]));
        atomicMin(&mm[1], __float_as_uint(s1[0]));
        atomicMax(&mm[2], __float_as_uint(s2[0]));
        atomicMax(&mm[3], __float_as_uint(s3[0]));
    }
}

// ---------------------------------------------------------------------------
// Kernel 2: edge MLP with WMMA + scatter-add
//   block = 256 threads = 8 waves; wave w handles batch b = w. The block
//   loops over NTILES tiles of 16 edges; each wave pushes its 16-row tile
//   through both MLP layers with v_wmma_f32_16x16x32_f16 and scatters +/- e
//   with fire-and-forget f32 atomics into the L2-resident agg buffer.
// ---------------------------------------------------------------------------
__global__ void __launch_bounds__(256)
k_edge(const float* __restrict__ x, const int* __restrict__ ei,
       const float* __restrict__ ea, const float* __restrict__ wmin,
       const float* __restrict__ wmax,
       const float* __restrict__ W1, const float* __restrict__ b1,
       const float* __restrict__ W2, const float* __restrict__ b2,
       const unsigned* __restrict__ mm, float* __restrict__ agg)
{
    __shared__ _Float16 sW1[EH][64];        // B-fragment layout: [n][k], k padded 35->64
    __shared__ _Float16 sW2[32][32];        // [n padded 22->32][k=32]
    __shared__ float    sb1[EH];
    __shared__ float    sb2[32];
    __shared__ _Float16 sFeat[8][16][FS];   // per-wave 16x64 feature tile (f16)
    __shared__ _Float16 sH[8][16][HS];      // per-wave 16x32 hidden tile (f16)
    __shared__ int      sIdx[8][2][16];     // per-wave src/tgt indices

    const int tid = threadIdx.x;

    // ---- stage weights into B-fragment-friendly LDS layout (once per block)
    for (int i = tid; i < EH * 64; i += 256) {
        int n = i >> 6, k = i & 63;
        sW1[n][k] = (k < 35) ? (_Float16)W1[k * EH + n] : (_Float16)0.0f;
    }
    for (int i = tid; i < 32 * 32; i += 256) {
        int n = i >> 5, k = i & 31;
        sW2[n][k] = (n < EOUT) ? (_Float16)W2[k * EOUT + n] : (_Float16)0.0f;
    }
    if (tid < EH) sb1[tid] = b1[tid];
    if (tid < 32) sb2[tid] = (tid < EOUT) ? b2[tid] : 0.0f;
    __syncthreads();

    const int w    = tid >> 5;     // wave id == batch id
    const int lane = tid & 31;
    const int m    = lane & 15;    // row within tile / N within fragment
    const int g    = lane >> 4;    // lane half-group
    const int b    = w;

    // ---- loop-invariant scalars
    const float wmn0 = wmin[0], wmn1 = wmin[1];
    const float wr0  = wmax[0] - wmn0, wr1 = wmax[1] - wmn1;
    const float mn0 = __uint_as_float(mm[0]), mn1 = __uint_as_float(mm[1]);
    const float mx0 = __uint_as_float(mm[2]), mx1 = __uint_as_float(mm[3]);
    const float inv0 = __builtin_amdgcn_rcpf(mx0 - mn0);
    const float inv1 = __builtin_amdgcn_rcpf(mx1 - mn1);
    const float l1b0 = sb1[m],      l1b1 = sb1[16 + m];
    const float l2b0 = sb2[m],      l2b1 = sb2[16 + m];

    // ---- zero this row's half of the (padded) feature tile once; the pad
    //      columns (35..63) are never overwritten by the loop below.
    {
        unsigned* zp = (unsigned*)&sFeat[w][m][g * 36];
#pragma unroll
        for (int i = 0; i < 18; ++i) zp[i] = 0u;
    }

    const _Float16* fr = &sFeat[w][m][0];
    const _Float16* hr = &sH[w][m][0];

#pragma unroll 1
    for (int tile = 0; tile < NTILES; ++tile) {
        const int e = (blockIdx.x * NTILES + tile) * 16 + m;

        const int sidx    = ei[e];
        const int tidx    = ei[NEDGES + e];
        const int nodeIdx = g ? tidx : sidx;
        sIdx[w][g][m] = nodeIdx;

        // ---- gather one node row (64B) — x fits in L2, gathers stay on-chip
        const float4* xr = (const float4*)(x + ((size_t)b * NNODES + nodeIdx) * INDIM);
        float4 f0 = xr[0], f1 = xr[1], f2 = xr[2], f3 = xr[3];
        float f[16] = { f0.x, f0.y, f0.z, f0.w,  f1.x, f1.y, f1.z, f1.w,
                        f2.x, f2.y, f2.z, f2.w,  f3.x, f3.y, f3.z, f3.w };

        // ---- write node features as packed f16 pairs (cols g*16 .. g*16+15)
        {
            unsigned* dst = (unsigned*)&sFeat[w][m][g * 16];
#pragma unroll
            for (int i = 0; i < 8; ++i) {
                union { _Float16 h[2]; unsigned u; } p;
                p.h[0] = (_Float16)f[2 * i];
                p.h[1] = (_Float16)f[2 * i + 1];
                dst[i] = p.u;
            }
        }

        // ---- edge scalar features (one lane per row: the src-half lane)
        if (g == 0) {
            float speed = f[14] * wr0 + wmn0;
            float direc = f[15] * wr1 + wmn1;
            float dist  = ea[2 * e + 0];
            float cdir  = ea[2 * e + 1];
            float ean0 = (dist - mn0) * inv0;
            float ean1 = (cdir - mn1) * inv1;
            float theta = fabsf(cdir - direc);
            float ew = fmaxf(speed * __cosf(theta) * __builtin_amdgcn_rcpf(dist), 0.0f);
            union { _Float16 h[2]; unsigned u; } p;
            p.h[0] = (_Float16)ean0;
            p.h[1] = (_Float16)ean1;
            *(unsigned*)&sFeat[w][m][32] = p.u;
            sFeat[w][m][34] = (_Float16)ew;
        }
        // branches converged: EXEC is all-ones for the WMMA block below.
        // Same-wave DS ops are in-order, so no barrier is needed before reload.

        // ---- layer 1: h16x32 = feat16x64(f16) @ W1t ; 2 K-tiles x 2 N-tiles
        v8f c0 = {}; v8f c1 = {};
#pragma unroll
        for (int kt = 0; kt < 2; ++kt) {
            // A fragment (16-bit A 16x32 layout): lane(m,g) holds
            // K = kt*32 + {g*8..g*8+7, 16+g*8..16+g*8+7}
            v8h lo = *(const v8h*)(fr + kt * 32 + g * 8);
            v8h hi = *(const v8h*)(fr + kt * 32 + 16 + g * 8);
            v16h a;
#pragma unroll
            for (int i = 0; i < 8; ++i) { a[i] = lo[i]; a[8 + i] = hi[i]; }
            // B fragments: lane(n=m,g) holds K = kt*32 + {g*8..g*8+7, 16+g*8..16+g*8+7}
            v16h bm0 = __builtin_shufflevector(*(const v8h*)(&sW1[m][kt * 32 + g * 8]), *(const v8h*)(&sW1[m][kt * 32 + 16 + g * 8]), 0,1,2,3,4,5,6,7,8,9,10,11,12,13,14,15);
            v16h bm1 = __builtin_shufflevector(*(const v8h*)(&sW1[16 + m][kt * 32 + g * 8]), *(const v8h*)(&sW1[16 + m][kt * 32 + 16 + g * 8]), 0,1,2,3,4,5,6,7,8,9,10,11,12,13,14,15);
            c0 = __builtin_amdgcn_wmma_f32_16x16x32_f16(false, a, false, bm0,
                                                        (short)0, c0, false, false);
            c1 = __builtin_amdgcn_wmma_f32_16x16x32_f16(false, a, false, bm1,
                                                        (short)0, c1, false, false);
        }

        // ---- bias + sigmoid on D fragments; round-trip h through LDS as f16
#pragma unroll
        for (int r = 0; r < 8; ++r) {
            int row = r + 8 * g;                 // D layout: M = r + 8*(lane/16)
            sH[w][row][m]      = (_Float16)sigmoidf_fast(c0[r] + l1b0);
            sH[w][row][16 + m] = (_Float16)sigmoidf_fast(c1[r] + l1b1);
        }

        // ---- layer 2: e16x22 = h16x32 @ W2t ; 1 K-tile x 2 N-tiles
        v8f d0 = {}; v8f d1 = {};
        {
            v8h lo = *(const v8h*)(hr + g * 8);
            v8h hi = *(const v8h*)(hr + 16 + g * 8);
            v16h a;
#pragma unroll
            for (int i = 0; i < 8; ++i) { a[i] = lo[i]; a[8 + i] = hi[i]; }
            v16h bm0 = __builtin_shufflevector(*(const v8h*)(&sW2[m][g * 8]), *(const v8h*)(&sW2[m][16 + g * 8]), 0,1,2,3,4,5,6,7,8,9,10,11,12,13,14,15);
            v16h bm1 = __builtin_shufflevector(*(const v8h*)(&sW2[16 + m][g * 8]), *(const v8h*)(&sW2[16 + m][16 + g * 8]), 0,1,2,3,4,5,6,7,8,9,10,11,12,13,14,15);
            d0 = __builtin_amdgcn_wmma_f32_16x16x32_f16(false, a, false, bm0,
                                                        (short)0, d0, false, false);
            d1 = __builtin_amdgcn_wmma_f32_16x16x32_f16(false, a, false, bm1,
                                                        (short)0, d1, false, false);
        }

        // ---- bias + sigmoid + scatter: agg[b,tgt] += e ; agg[b,src] -= e
#pragma unroll
        for (int r = 0; r < 8; ++r) {
            int row = r + 8 * g;
            int tg  = sIdx[w][1][row];
            int sr  = sIdx[w][0][row];
            float* at = agg + ((size_t)b * NNODES + tg) * EOUT;
            float* as = agg + ((size_t)b * NNODES + sr) * EOUT;
            float v0 = sigmoidf_fast(d0[r] + l2b0);   // column m (< 16 < 22)
            atomicAdd(at + m, v0);
            atomicAdd(as + m, -v0);
            if (m < 6) {                              // columns 16..21
                float v1 = sigmoidf_fast(d1[r] + l2b1);
                atomicAdd(at + 16 + m, v1);
                atomicAdd(as + 16 + m, -v1);
            }
        }
    }
}

// ---------------------------------------------------------------------------
// Kernel 3: node MLP out = sigmoid(agg @ Wn + bn)  (tiny: 56 MFLOP)
// ---------------------------------------------------------------------------
__global__ void k_node(const float* __restrict__ agg, const float* __restrict__ Wn,
                       const float* __restrict__ bn, float* __restrict__ out, int total) {
    __shared__ float sWn[EOUT][NOUT];
    __shared__ float sbn[NOUT];
    for (int i = threadIdx.x; i < EOUT * NOUT; i += blockDim.x)
        sWn[i / NOUT][i % NOUT] = Wn[i];
    if (threadIdx.x < NOUT) sbn[threadIdx.x] = bn[threadIdx.x];
    __syncthreads();

    int id = blockIdx.x * blockDim.x + threadIdx.x;
    if (id >= total) return;
    const float* ar = agg + (size_t)id * EOUT;
    float a[EOUT];
#pragma unroll
    for (int k = 0; k < EOUT; ++k) a[k] = ar[k];
    float* orow = out + (size_t)id * NOUT;
#pragma unroll
    for (int j = 0; j < NOUT; ++j) {
        float acc = sbn[j];
#pragma unroll
        for (int k = 0; k < EOUT; ++k) acc = fmaf(a[k], sWn[k][j], acc);
        orow[j] = sigmoidf_fast(acc);
    }
}

// ---------------------------------------------------------------------------
extern "C" void kernel_launch(void* const* d_in, const int* in_sizes, int n_in,
                              void* d_out, int out_size, void* d_ws, size_t ws_size,
                              hipStream_t stream) {
    const float* x    = (const float*)d_in[0];
    const int*   ei   = (const int*)d_in[1];
    const float* ea   = (const float*)d_in[2];
    const float* wmin = (const float*)d_in[3];
    const float* wmax = (const float*)d_in[4];
    const float* W1   = (const float*)d_in[5];
    const float* b1   = (const float*)d_in[6];
    const float* W2   = (const float*)d_in[7];
    const float* b2   = (const float*)d_in[8];
    const float* Wn   = (const float*)d_in[9];
    const float* bn   = (const float*)d_in[10];
    float* out = (float*)d_out;

    // workspace layout: [0..15] min/max uint slots, [64..] agg (8*10000*22 f32)
    unsigned* mm = (unsigned*)d_ws;
    float* agg   = (float*)((char*)d_ws + 64);
    const int aggN = NBATCH * NNODES * EOUT;

    k_init  <<<512, 256, 0, stream>>>(mm, agg, aggN);
    k_minmax<<<512, 256, 0, stream>>>(ea, mm, NEDGES);
    k_edge  <<<NEDGES / (16 * NTILES), 256, 0, stream>>>(x, ei, ea, wmin, wmax,
                                                         W1, b1, W2, b2, mm, agg);
    k_node  <<<(NBATCH * NNODES + 255) / 256, 256, 0, stream>>>(agg, Wn, bn, out,
                                                                NBATCH * NNODES);
}
